// DynamicSparseAttention_74577812127939
// MI455X (gfx1250) — matrix operand order repaired and checked by reading, not compiled, not run on hardware
//
#include <hip/hip_runtime.h>
#include <hip/hip_bf16.h>

// ---------------------------------------------------------------------------
// Problem constants (match reference)
// ---------------------------------------------------------------------------
#define BATCH    2
#define SEQ      2048
#define N_EMBD   2048
#define N_HEAD   16
#define HEAD_DIM 128
#define GATE_HID 64
#define BT       (BATCH * SEQ)      // 4096 rows
#define QKV_N    (3 * N_EMBD)       // 6144

typedef __attribute__((ext_vector_type(16))) __bf16 bf16x16;
typedef __attribute__((ext_vector_type(8)))  __bf16 bf16x8;
typedef __attribute__((ext_vector_type(8)))  float  f32x8;

// ---------------------------------------------------------------------------
// WMMA helpers (CDNA5 16x16x32 bf16, fp32 accumulate)
// ---------------------------------------------------------------------------
__device__ __forceinline__ f32x8 wmma_bf16(bf16x16 a, bf16x16 b, f32x8 c) {
  return __builtin_amdgcn_wmma_f32_16x16x32_bf16(false, a, false, b,
                                                 (short)0, c, false, false);
}

// A-operand (16x32, M x K), source row-major [M rows, ld elems apart].
// ISA layout: lane L -> row m = L%16; elems 0..7 : K = (L/16)*8 + i,
//             elems 8..15: K = 16 + (L/16)*8 + (i-8)  -> two contiguous runs.
__device__ __forceinline__ bf16x16 load_a_frag(const __bf16* base, int ld) {
  const int lane = threadIdx.x & 31;
  const int hlf  = lane >> 4;
  const int m    = lane & 15;
  const __bf16* p = base + (size_t)m * ld + hlf * 8;
  bf16x8 lo = *(const bf16x8*)(p);
  bf16x8 hi = *(const bf16x8*)(p + 16);
  return __builtin_shufflevector(lo, hi, 0, 1, 2, 3, 4, 5, 6, 7,
                                 8, 9, 10, 11, 12, 13, 14, 15);
}

// B-operand (32x16, K x N), source = "Bt" row-major [N rows, ld elems apart],
// i.e. B(k,n) = Bt[n][k].  ISA layout: lane L -> col n = L%16;
// elems 0..7 : K = (L/16)*8 + i, elems 8..15: K = 16 + (L/16)*8 + (i-8).
__device__ __forceinline__ bf16x16 load_b_frag(const __bf16* base, int ld) {
  const int lane = threadIdx.x & 31;
  const int hlf  = lane >> 4;
  const int n    = lane & 15;
  const __bf16* p = base + (size_t)n * ld + hlf * 8;
  bf16x8 lo = *(const bf16x8*)(p);
  bf16x8 hi = *(const bf16x8*)(p + 16);
  return __builtin_shufflevector(lo, hi, 0, 1, 2, 3, 4, 5, 6, 7,
                                 8, 9, 10, 11, 12, 13, 14, 15);
}

// ---------------------------------------------------------------------------
// Prep kernels: fp32 -> bf16 convert, and fp32 [K,N] -> bf16 [N,K] transpose
// ---------------------------------------------------------------------------
__global__ void f32_to_bf16_kernel(const float* __restrict__ in,
                                   __bf16* __restrict__ out, int n) {
  int i = blockIdx.x * blockDim.x + threadIdx.x;
  int stride = gridDim.x * blockDim.x;
  for (; i < n; i += stride) out[i] = (__bf16)in[i];
}

__global__ __launch_bounds__(256)
void transpose_f32_to_bf16_kernel(const float* __restrict__ W,
                                  __bf16* __restrict__ Wt, int K, int N) {
  __shared__ float tile[32][33];
  const int n0 = blockIdx.x * 32;
  const int k0 = blockIdx.y * 32;
  for (int r = threadIdx.y; r < 32; r += 8)
    tile[r][threadIdx.x] = W[(size_t)(k0 + r) * N + n0 + threadIdx.x];
  __syncthreads();
  for (int r = threadIdx.y; r < 32; r += 8)
    Wt[(size_t)(n0 + r) * K + k0 + threadIdx.x] = (__bf16)tile[threadIdx.x][r];
}

// ---------------------------------------------------------------------------
// GEMM: C[M,N] = A[M,K](bf16) * Bt[N,K]^T(bf16) + bias
// MODE 0: fp32 row-major out.  MODE 1: scatter epilogue into q/k/vt (bf16).
// Block = 256 threads (8 waves); block tile 128x128; wave tile 16x128.
// ---------------------------------------------------------------------------
template <int MODE>
__global__ __launch_bounds__(256)
void gemm_bf16_wmma_kernel(const __bf16* __restrict__ A,
                           const __bf16* __restrict__ Bt,
                           const float* __restrict__ bias,
                           float* __restrict__ outF,
                           __bf16* __restrict__ qo,
                           __bf16* __restrict__ ko,
                           __bf16* __restrict__ vto,
                           int M, int N, int K) {
  const int wave  = threadIdx.x >> 5;
  const int lane  = threadIdx.x & 31;
  const int hlf   = lane >> 4;
  const int ln16  = lane & 15;
  const int mBase = blockIdx.y * 128 + wave * 16;
  const int nBase = blockIdx.x * 128;

  f32x8 acc[8] = {};
  const __bf16* aRow = A + (size_t)mBase * K;

  for (int kc = 0; kc < K; kc += 32) {
    __builtin_prefetch(aRow + kc + 32, 0, 0);  // global_prefetch_b8
    bf16x16 a = load_a_frag(aRow + kc, K);
#pragma unroll
    for (int nt = 0; nt < 8; ++nt) {
      bf16x16 b = load_b_frag(Bt + (size_t)(nBase + nt * 16) * K + kc, K);
      acc[nt] = wmma_bf16(a, b, acc[nt]);
    }
  }

#pragma unroll
  for (int nt = 0; nt < 8; ++nt) {
    const int col = nBase + nt * 16 + ln16;
    const float bs = bias[col];
#pragma unroll
    for (int j = 0; j < 8; ++j) {
      const int row = mBase + j + 8 * hlf;
      const float v = acc[nt][j] + bs;
      if (MODE == 0) {
        outF[(size_t)row * N + col] = v;
      } else {
        const int which = col >> 11;       // 0:q 1:k 2:v  (col / 2048)
        const int rem   = col & 2047;
        const int h     = rem >> 7;        // head
        const int d     = rem & 127;       // dim
        const int bidx  = row >> 11;       // row / SEQ
        const int t     = row & 2047;
        const int bh    = bidx * N_HEAD + h;
        const __bf16 bv = (__bf16)v;
        if (which == 0)
          qo[((size_t)bh * SEQ + t) * HEAD_DIM + d] = bv;
        else if (which == 1)
          ko[((size_t)bh * SEQ + t) * HEAD_DIM + d] = bv;
        else  // v stored transposed: [B,H,D,T] for contiguous PV B-operand
          vto[((size_t)bh * HEAD_DIM + d) * SEQ + t] = bv;
      }
    }
  }
}

// ---------------------------------------------------------------------------
// Gate kernel: one block per (b,h). mean_t q -> relu(.@Wg1+b) -> sigmoid(.@Wg2+b)
// ---------------------------------------------------------------------------
__global__ __launch_bounds__(128)
void gate_kernel(const __bf16* __restrict__ q,
                 const float* __restrict__ Wg1, const float* __restrict__ bg1,
                 const float* __restrict__ Wg2, const float* __restrict__ bg2,
                 float* __restrict__ gates) {
  const int bh = blockIdx.x;
  const int d  = threadIdx.x;  // 0..127
  const __bf16* Q = q + (size_t)bh * SEQ * HEAD_DIM;

  float s = 0.f;
  for (int t = 0; t < SEQ; ++t) s += (float)Q[(size_t)t * HEAD_DIM + d];
  const float qm = s * (1.0f / SEQ);

  __shared__ float sh_qm[HEAD_DIM];
  __shared__ float sh_h[GATE_HID];
  sh_qm[d] = qm;
  __syncthreads();

  if (d < GATE_HID) {
    float a = bg1[d];
    for (int i = 0; i < HEAD_DIM; ++i) a += sh_qm[i] * Wg1[i * GATE_HID + d];
    sh_h[d] = fmaxf(a, 0.f);
  }
  __syncthreads();

  if (d == 0) {
    float a = bg2[0];
    for (int i = 0; i < GATE_HID; ++i) a += sh_h[i] * Wg2[i];
    gates[bh] = 1.0f / (1.0f + __expf(-a));
  }
}

// ---------------------------------------------------------------------------
// Flash attention: block = 4 waves, each wave owns 16 query rows of one (b,h).
// S = Q K^T (WMMA), online softmax (exp2, shfl reductions), P via LDS
// C-layout -> A-layout, Y += P V (WMMA vs transposed V). Gate in epilogue.
// ---------------------------------------------------------------------------
__global__ __launch_bounds__(128)
void attn_wmma_kernel(const __bf16* __restrict__ q,
                      const __bf16* __restrict__ kk,
                      const __bf16* __restrict__ vt,
                      const float* __restrict__ gates,
                      __bf16* __restrict__ y) {
  __shared__ __bf16 pLds[4 * 16 * 32];  // one 16x32 P tile per wave

  const int wave = threadIdx.x >> 5;
  const int lane = threadIdx.x & 31;
  const int hlf  = lane >> 4;
  const int ln16 = lane & 15;

  const int ntq   = SEQ / 64;  // query tiles per (b,h)
  const int bh    = blockIdx.x / ntq;
  const int qt    = blockIdx.x % ntq;
  const int mBase = qt * 64 + wave * 16;

  const __bf16* Q  = q  + (size_t)bh * SEQ * HEAD_DIM;
  const __bf16* Kp = kk + (size_t)bh * SEQ * HEAD_DIM;
  const __bf16* Vt = vt + (size_t)bh * HEAD_DIM * SEQ;

  bf16x16 qa[4];
#pragma unroll
  for (int c = 0; c < 4; ++c)
    qa[c] = load_a_frag(Q + (size_t)mBase * HEAD_DIM + c * 32, HEAD_DIM);

  // 1/sqrt(128) * log2(e): scale once, use exp2
  const float sl = 0.08838834764831845f * 1.4426950408889634f;
  float mi[8], li[8], corr[8];
#pragma unroll
  for (int j = 0; j < 8; ++j) { mi[j] = -1e30f; li[j] = 0.f; }
  f32x8 acc[8] = {};

  __bf16* myP = pLds + wave * (16 * 32);

  for (int kt = 0; kt < SEQ; kt += 32) {
    const __bf16* Kb = Kp + (size_t)kt * HEAD_DIM;
    f32x8 S0 = {}, S1 = {};
#pragma unroll
    for (int c = 0; c < 4; ++c) {
      bf16x16 b0 = load_b_frag(Kb, HEAD_DIM);                  // keys kt..kt+15
      bf16x16 b1 = load_b_frag(Kb + 16 * HEAD_DIM, HEAD_DIM);  // keys +16..+31
      S0 = wmma_bf16(qa[c], b0, S0);
      S1 = wmma_bf16(qa[c], b1, S1);
      Kb += 32;  // next 32-dim chunk of D
    }

    // online softmax; C-layout: lane holds col n=ln16, rows m = j + 8*hlf,
    // so xor-masks 1/2/4/8 reduce across the 16 lanes holding one row.
#pragma unroll
    for (int j = 0; j < 8; ++j) {
      float s0 = S0[j] * sl, s1 = S1[j] * sl;
      float mx = fmaxf(s0, s1);
      mx = fmaxf(mx, __shfl_xor(mx, 1, 32));
      mx = fmaxf(mx, __shfl_xor(mx, 2, 32));
      mx = fmaxf(mx, __shfl_xor(mx, 4, 32));
      mx = fmaxf(mx, __shfl_xor(mx, 8, 32));
      const float mn = fmaxf(mi[j], mx);
      const float c0 = exp2f(mi[j] - mn);
      mi[j] = mn;
      const float p0 = exp2f(s0 - mn);
      const float p1 = exp2f(s1 - mn);
      float rs = p0 + p1;
      rs += __shfl_xor(rs, 1, 32);
      rs += __shfl_xor(rs, 2, 32);
      rs += __shfl_xor(rs, 4, 32);
      rs += __shfl_xor(rs, 8, 32);
      li[j] = li[j] * c0 + rs;
      corr[j] = c0;
      S0[j] = p0;
      S1[j] = p1;
    }
#pragma unroll
    for (int nt = 0; nt < 8; ++nt)
#pragma unroll
      for (int j = 0; j < 8; ++j) acc[nt][j] *= corr[j];

    // P tile (C-layout) -> LDS row-major [16][32] (wave-private, no barrier;
    // LDS ops are in-order within a wave)
#pragma unroll
    for (int j = 0; j < 8; ++j) {
      const int m = j + 8 * hlf;
      myP[m * 32 + ln16]      = (__bf16)S0[j];
      myP[m * 32 + 16 + ln16] = (__bf16)S1[j];
    }
    const bf16x16 pa = load_a_frag(myP, 32);  // ds_load back in A-layout

#pragma unroll
    for (int nt = 0; nt < 8; ++nt) {
      // B(k=key, n=d) = V[key][d] = Vt[d][key]; row n = d, contiguous keys.
      bf16x16 vb = load_b_frag(Vt + (size_t)(nt * 16) * SEQ + kt, SEQ);
      acc[nt] = wmma_bf16(pa, vb, acc[nt]);
    }
  }

  const float g = gates[bh];
  const int b = bh / N_HEAD;
  const int h = bh % N_HEAD;
#pragma unroll
  for (int nt = 0; nt < 8; ++nt) {
#pragma unroll
    for (int j = 0; j < 8; ++j) {
      const int t = mBase + j + 8 * hlf;
      const int d = nt * 16 + ln16;
      const float v = acc[nt][j] / li[j] * g;
      y[((size_t)(b * SEQ + t)) * N_EMBD + h * HEAD_DIM + d] = (__bf16)v;
    }
  }
}

// ---------------------------------------------------------------------------
// Host-side launch
// ---------------------------------------------------------------------------
extern "C" void kernel_launch(void* const* d_in, const int* in_sizes, int n_in,
                              void* d_out, int out_size, void* d_ws,
                              size_t ws_size, hipStream_t stream) {
  const float* x      = (const float*)d_in[0];
  const float* W_attn = (const float*)d_in[1];
  const float* b_attn = (const float*)d_in[2];
  const float* W_proj = (const float*)d_in[3];
  const float* b_proj = (const float*)d_in[4];
  const float* W_g1   = (const float*)d_in[5];
  const float* b_g1   = (const float*)d_in[6];
  const float* W_g2   = (const float*)d_in[7];
  const float* b_g2   = (const float*)d_in[8];

  float* out   = (float*)d_out;
  float* gates = out + (size_t)BT * N_EMBD;  // 32 gate outputs at the tail

  // workspace carve-up (256B aligned)
  size_t off = 0;
  auto alloc = [&](size_t bytes) -> void* {
    void* p = (char*)d_ws + off;
    off += (bytes + 255) & ~(size_t)255;
    return p;
  };
  __bf16* xb     = (__bf16*)alloc((size_t)BT * N_EMBD * 2);
  __bf16* WattnT = (__bf16*)alloc((size_t)QKV_N * N_EMBD * 2);
  __bf16* WprojT = (__bf16*)alloc((size_t)N_EMBD * N_EMBD * 2);
  __bf16* qb     = (__bf16*)alloc((size_t)BT * N_EMBD * 2);
  __bf16* kb     = (__bf16*)alloc((size_t)BT * N_EMBD * 2);
  __bf16* vtb    = (__bf16*)alloc((size_t)BT * N_EMBD * 2);
  __bf16* yb     = (__bf16*)alloc((size_t)BT * N_EMBD * 2);
  (void)ws_size;

  // 1) precision prep
  f32_to_bf16_kernel<<<4096, 256, 0, stream>>>(x, xb, BT * N_EMBD);
  transpose_f32_to_bf16_kernel<<<dim3(QKV_N / 32, N_EMBD / 32), dim3(32, 8), 0,
                                 stream>>>(W_attn, WattnT, N_EMBD, QKV_N);
  transpose_f32_to_bf16_kernel<<<dim3(N_EMBD / 32, N_EMBD / 32), dim3(32, 8), 0,
                                 stream>>>(W_proj, WprojT, N_EMBD, N_EMBD);

  // 2) QKV projection with scatter into q / k / v^T
  gemm_bf16_wmma_kernel<1><<<dim3(QKV_N / 128, BT / 128), 256, 0, stream>>>(
      xb, WattnT, b_attn, nullptr, qb, kb, vtb, BT, QKV_N, N_EMBD);

  // 3) per-head gates (written into tail of d_out)
  gate_kernel<<<BATCH * N_HEAD, 128, 0, stream>>>(qb, W_g1, b_g1, W_g2, b_g2,
                                                  gates);

  // 4) flash attention (gated), bf16 y in [B,T,C]
  attn_wmma_kernel<<<BATCH * N_HEAD * (SEQ / 64), 128, 0, stream>>>(
      qb, kb, vtb, gates, yb);

  // 5) output projection -> fp32 d_out
  gemm_bf16_wmma_kernel<0><<<dim3(N_EMBD / 128, BT / 128), 256, 0, stream>>>(
      yb, WprojT, b_proj, out, nullptr, nullptr, nullptr, BT, N_EMBD, N_EMBD);
}
